// PicoSleepNetBaseline_23038204575757
// MI455X (gfx1250) — matrix operand order repaired and checked by reading, not compiled, not run on hardware
//
#include <hip/hip_runtime.h>

// ---------------------------------------------------------------------------
// PicoSleepNet forward, fused SNN scan for gfx1250 (MI455X), bf16 WMMA.
// v2: software-pipelined LDS fragment loads (double-buffered, chunk j+2
//     issued behind WMMA j), register double-buffered x prefetch (HBM latency
//     hidden behind compute), double-buffered spike LDS -> 2 barriers/step.
// ---------------------------------------------------------------------------

typedef __attribute__((ext_vector_type(16))) __bf16 v16bf;
typedef __attribute__((ext_vector_type(8)))  float  v8f;

#define TB      64      // batch rows per block
#define RP      160     // padded R (150)
#define DP      96      // padded D (80)
#define HP      64      // padded H (50)
#define T_STEPS 40
#define R_REAL  150
#define D_REAL  80
#define H_REAL  50
#define C_OUT   5
#define TAU_F   0.95f
#define VTH_F   1.0f
#define QMAX    31.0f   // 6-bit signed symmetric

// ws layout (bytes)
#define WS_WIN   0                         // bf16 [160][96]  = 30720
#define WS_WREC  30720                     // bf16 [160][160] = 51200
#define WS_WHID  81920                     // bf16 [64][160]  = 20480
#define WS_WOUT  102400                    // f32  [5][50]
#define WS_BOUT  103424                    // f32  [5]

// LDS layout (bytes), all 16B aligned
#define L_WIN   0                          // 30720
#define L_WREC  30720                      // 51200
#define L_WHID  81920                      // 20480
#define L_X     102400                     // 12288
#define L_S0    114688                     // 20480
#define L_S1    135168                     // 20480
#define L_ACC   L_X                        // 16384, aliases sX+sS0 (post-scan)
#define LDS_BYTES 155648                   // 152KB -> 2 blocks/WGP (304<=320KB)

// ---- fragment loaders (contiguous k order, same for A and B) ----------
// A (16x32 bf16): lane (lo,hi); row = m0+lo.
//   elem e -> K = k0 + hi*16 + e
// Precomputed lane pointer: S + (m0+lo)*ld + k0 + hi*16 ; second uint4 at +8.
__device__ __forceinline__ v16bf ldA(const __bf16* p) {
  union { v16bf v; uint4 u[2]; } f;
  f.u[0] = *(const uint4*)p;
  f.u[1] = *(const uint4*)(p + 8);
  return f.v;
}
// B (32x16 bf16): elem e -> K = k0 + hi*16 + e, N = n0 + lo.
// Lane pointer: W + (n0+lo)*ld + k0 + hi*16 ; second uint4 at +8.
__device__ __forceinline__ v16bf ldB(const __bf16* p) {
  union { v16bf v; uint4 u[2]; } f;
  f.u[0] = *(const uint4*)p;
  f.u[1] = *(const uint4*)(p + 8);
  return f.v;
}

__device__ __forceinline__ v8f wmma_bf16(v16bf a, v16bf b, v8f c) {
  return __builtin_amdgcn_wmma_f32_16x16x32_bf16(
      false, a, false, b, (short)0, c, false, false);
}

__device__ __forceinline__ float fq(float v, float scale) {
  return fminf(fmaxf(rintf(v / scale), -QMAX), QMAX) * scale;
}

// ---------------------------------------------------------------------------
// Prep: masked + 6-bit fake-quantized weights, padded, bf16 (Wout/bout f32).
// blockIdx.x: 0=Win 1=Wrec 2=Whid 3=Wout(+bout)
// ---------------------------------------------------------------------------
__global__ __launch_bounds__(256) void snn_prep(
    const float* __restrict__ w_in,  const float* __restrict__ m_in,
    const float* __restrict__ w_rec, const float* __restrict__ m_rec,
    const float* __restrict__ w_hid, const float* __restrict__ m_hid,
    const float* __restrict__ w_out, const float* __restrict__ m_out,
    const float* __restrict__ b_out,
    __bf16* __restrict__ oWin, __bf16* __restrict__ oWrec,
    __bf16* __restrict__ oWhid, float* __restrict__ oWout,
    float* __restrict__ oBout) {
  __shared__ float red[256];
  const int tid = threadIdx.x;
  const int mat = blockIdx.x;

  const float *w, *m; int n;
  if      (mat == 0) { w = w_in;  m = m_in;  n = R_REAL * D_REAL; }
  else if (mat == 1) { w = w_rec; m = m_rec; n = R_REAL * R_REAL; }
  else if (mat == 2) { w = w_hid; m = m_hid; n = H_REAL * R_REAL; }
  else               { w = w_out; m = m_out; n = C_OUT * H_REAL; }

  float mx = 0.0f;
  for (int i = tid; i < n; i += 256) {
    float v = (m[i] >= 0.0f) ? w[i] : 0.0f;
    mx = fmaxf(mx, fabsf(v));
  }
  red[tid] = mx;
  __syncthreads();
  for (int s = 128; s > 0; s >>= 1) {
    if (tid < s) red[tid] = fmaxf(red[tid], red[tid + s]);
    __syncthreads();
  }
  const float scale = fmaxf(red[0], 1e-8f) / QMAX;

  if (mat == 0) {
    for (int i = tid; i < RP * DP; i += 256) {
      int r = i / DP, d = i % DP;
      float q = 0.0f;
      if (r < R_REAL && d < D_REAL) {
        int s0 = r * D_REAL + d;
        float v = (m_in[s0] >= 0.0f) ? w_in[s0] : 0.0f;
        q = fq(v, scale);
      }
      oWin[i] = (__bf16)q;
    }
  } else if (mat == 1) {
    for (int i = tid; i < RP * RP; i += 256) {
      int r = i / RP, k = i % RP;
      float q = 0.0f;
      if (r < R_REAL && k < R_REAL) {
        int s0 = r * R_REAL + k;
        float v = (m_rec[s0] >= 0.0f) ? w_rec[s0] : 0.0f;
        q = fq(v, scale);
      }
      oWrec[i] = (__bf16)q;
    }
  } else if (mat == 2) {
    for (int i = tid; i < HP * RP; i += 256) {
      int h = i / RP, k = i % RP;
      float q = 0.0f;
      if (h < H_REAL && k < R_REAL) {
        int s0 = h * R_REAL + k;
        float v = (m_hid[s0] >= 0.0f) ? w_hid[s0] : 0.0f;
        q = fq(v, scale);
      }
      oWhid[i] = (__bf16)q;
    }
  } else {
    for (int i = tid; i < C_OUT * H_REAL; i += 256) {
      float v = (m_out[i] >= 0.0f) ? w_out[i] : 0.0f;
      oWout[i] = fq(v, scale);
    }
    if (tid == 0) {
      float bm = 0.0f;
      for (int c = 0; c < C_OUT; c++) bm = fmaxf(bm, fabsf(b_out[c]));
      float bs = fmaxf(bm, 1e-8f) / QMAX;
      for (int c = 0; c < C_OUT; c++) oBout[c] = fq(b_out[c], bs);
    }
  }
}

// ---------------------------------------------------------------------------
// Main fused scan: 256 blocks x 256 threads (8 wave32). TB=64 batch rows.
// Wave w owns res tiles {w, w+8, ..., w+32} of 4x10, hid tiles {2w, 2w+1}
// of 4x4. v_res/v_hid/acc live in registers across all 40 steps.
// ---------------------------------------------------------------------------
__global__ __launch_bounds__(256) void snn_main(
    const float* __restrict__ x,
    const float* __restrict__ bias_res,
    const float* __restrict__ bias_hid,
    const __bf16* __restrict__ gWin,
    const __bf16* __restrict__ gWrec,
    const __bf16* __restrict__ gWhid,
    const float* __restrict__ gWout,
    const float* __restrict__ gBout,
    float* __restrict__ out) {
  extern __shared__ char smem[];
  __bf16* sWin  = (__bf16*)(smem + L_WIN);    // [160][96]
  __bf16* sWrec = (__bf16*)(smem + L_WREC);   // [160][160]
  __bf16* sWhid = (__bf16*)(smem + L_WHID);   // [64][160]
  __bf16* sX    = (__bf16*)(smem + L_X);      // [64][96]
  __bf16* sS0   = (__bf16*)(smem + L_S0);     // [64][160] spikes, parity 0
  __bf16* sS1   = (__bf16*)(smem + L_S1);     // [64][160] spikes, parity 1
  float*  sAcc  = (float*) (smem + L_ACC);    // [64][64] (aliases sX+sS0)

  const int tid  = threadIdx.x;
  const int lane = tid & 31;
  const int wv   = tid >> 5;
  const int lo   = lane & 15;
  const int hi   = lane >> 4;
  const int b0   = blockIdx.x * TB;

  // --- weights -> LDS (uint4 copies) ---
  {
    const uint4* s0 = (const uint4*)gWin;  uint4* d0 = (uint4*)sWin;
    for (int i = tid; i < RP * DP * 2 / 16; i += 256) d0[i] = s0[i];
    const uint4* s1 = (const uint4*)gWrec; uint4* d1 = (uint4*)sWrec;
    for (int i = tid; i < RP * RP * 2 / 16; i += 256) d1[i] = s1[i];
    const uint4* s2 = (const uint4*)gWhid; uint4* d2 = (uint4*)sWhid;
    for (int i = tid; i < HP * RP * 2 / 16; i += 256) d2[i] = s2[i];
  }
  // initial spikes (parity-0 buffer) = 0; sS1 is fully written in step 0
  for (int i = tid; i < TB * RP; i += 256) sS0[i] = (__bf16)0.0f;
  // zero the x pad columns d in [80,96) once; never rewritten
  for (int i = tid; i < TB * (DP - D_REAL); i += 256) {
    int r = i / (DP - D_REAL), d = D_REAL + i % (DP - D_REAL);
    sX[r * DP + d] = (__bf16)0.0f;
  }

  // --- x prefetch mapping: 5 float4 groups/thread = 64 rows x 20 groups ---
  int xr_row[5], xr_dq[5];
#pragma unroll
  for (int j = 0; j < 5; j++) {
    int g = tid + 256 * j;
    xr_row[j] = g / (D_REAL / 4);
    xr_dq[j]  = (g % (D_REAL / 4)) * 4;
  }
  // preload x[:, 0, :] and store to sX
  float4 xr[5];
#pragma unroll
  for (int j = 0; j < 5; j++)
    xr[j] = *(const float4*)(x + (((size_t)(b0 + xr_row[j])) * T_STEPS + 0)
                                 * D_REAL + xr_dq[j]);
#pragma unroll
  for (int j = 0; j < 5; j++) {
    union { __bf16 h[4]; uint2 u; } pk;
    pk.h[0] = (__bf16)xr[j].x; pk.h[1] = (__bf16)xr[j].y;
    pk.h[2] = (__bf16)xr[j].z; pk.h[3] = (__bf16)xr[j].w;
    *(uint2*)(sX + xr_row[j] * DP + xr_dq[j]) = pk.u;
  }

  // --- register state ---
  float vres[5][8];
  float vhid[2][8], acch[2][8];
  float bres[5], bhid[2];
  int resM[5], resN[5], hidM[2], hidN[2];
#pragma unroll
  for (int i = 0; i < 5; i++) {
    int tr = wv + 8 * i;
    resM[i] = (tr / 10) * 16;
    resN[i] = (tr % 10) * 16;
    int ng = resN[i] + lo;
    bres[i] = (ng < R_REAL) ? bias_res[ng] : 0.0f;
#pragma unroll
    for (int r = 0; r < 8; r++) vres[i][r] = 0.0f;
  }
#pragma unroll
  for (int i = 0; i < 2; i++) {
    int th = wv * 2 + i;
    hidM[i] = (th / 4) * 16;
    hidN[i] = (th % 4) * 16;
    int ng = hidN[i] + lo;
    bhid[i] = (ng < H_REAL) ? bias_hid[ng] : 0.0f;
#pragma unroll
    for (int r = 0; r < 8; r++) { vhid[i][r] = 0.0f; acch[i][r] = 0.0f; }
  }
  __syncthreads();

#pragma unroll 1
  for (int t = 0; t < T_STEPS; t++) {
    const __bf16* sSrd = (t & 1) ? sS1 : sS0;   // spikes from step t-1
    __bf16*       sSwr = (t & 1) ? sS0 : sS1;   // spikes of step t

    // ---- issue global prefetch of x[:, t+1, :] (consumed after barrier) ----
    float4 xn[5];
    const bool more = (t + 1 < T_STEPS);
    if (more) {
#pragma unroll
      for (int j = 0; j < 5; j++)
        xn[j] = *(const float4*)(x + (((size_t)(b0 + xr_row[j])) * T_STEPS
                                      + (t + 1)) * D_REAL + xr_dq[j]);
    }

    // ---- reservoir: v = TAU*v + x@Win^T + s_prev@Wrec^T + b; spike ----
#pragma unroll
    for (int i = 0; i < 5; i++) {
      const int m0 = resM[i], n0 = resN[i];
      const __bf16* aP[8]; const __bf16* bP[8];
#pragma unroll
      for (int j = 0; j < 3; j++) {
        aP[j] = sX + (m0 + lo) * DP + j * 32 + hi * 16;
        bP[j] = sWin + (n0 + lo) * DP + j * 32 + hi * 16;
      }
#pragma unroll
      for (int j = 0; j < 5; j++) {
        aP[3 + j] = sSrd + (m0 + lo) * RP + j * 32 + hi * 16;
        bP[3 + j] = sWrec + (n0 + lo) * RP + j * 32 + hi * 16;
      }
      v8f acc = {0.f, 0.f, 0.f, 0.f, 0.f, 0.f, 0.f, 0.f};
      v16bf abuf[2], bbuf[2];
      abuf[0] = ldA(aP[0]); bbuf[0] = ldB(bP[0]);
      abuf[1] = ldA(aP[1]); bbuf[1] = ldB(bP[1]);
#pragma unroll
      for (int j = 0; j < 8; j++) {          // pipelined: wmma j, load j+2
        acc = wmma_bf16(abuf[j & 1], bbuf[j & 1], acc);
        if (j + 2 < 8) { abuf[j & 1] = ldA(aP[j + 2]);
                         bbuf[j & 1] = ldB(bP[j + 2]); }
      }
      // LIF update + immediate spike store (different buffer than sSrd)
#pragma unroll
      for (int r = 0; r < 8; r++) {
        float v = TAU_F * vres[i][r] + acc[r] + bres[i];
        float s = (v - VTH_F > 0.0f) ? 1.0f : 0.0f;
        vres[i][r] = v - s * VTH_F;
        sSwr[(m0 + hi * 8 + r) * RP + n0 + lo] = (__bf16)s;
      }
    }
    __syncthreads();  // spikes visible; sX / sSrd reads complete

    // ---- hidden layer: v = TAU*v + s@Whid^T + b; spike; accumulate ----
#pragma unroll
    for (int i = 0; i < 2; i++) {
      const int m0 = hidM[i], n0 = hidN[i];
      const __bf16* aP[5]; const __bf16* bP[5];
#pragma unroll
      for (int j = 0; j < 5; j++) {
        aP[j] = sSwr + (m0 + lo) * RP + j * 32 + hi * 16;
        bP[j] = sWhid + (n0 + lo) * RP + j * 32 + hi * 16;
      }
      v8f acc = {0.f, 0.f, 0.f, 0.f, 0.f, 0.f, 0.f, 0.f};
      v16bf abuf[2], bbuf[2];
      abuf[0] = ldA(aP[0]); bbuf[0] = ldB(bP[0]);
      abuf[1] = ldA(aP[1]); bbuf[1] = ldB(bP[1]);
#pragma unroll
      for (int j = 0; j < 5; j++) {
        acc = wmma_bf16(abuf[j & 1], bbuf[j & 1], acc);
        if (j + 2 < 5) { abuf[j & 1] = ldA(aP[j + 2]);
                         bbuf[j & 1] = ldB(bP[j + 2]); }
      }
#pragma unroll
      for (int r = 0; r < 8; r++) {
        float v = TAU_F * vhid[i][r] + acc[r] + bhid[i];
        float s = (v - VTH_F > 0.0f) ? 1.0f : 0.0f;
        vhid[i][r] = v - s * VTH_F;
        acch[i][r] += s;
      }
    }

    // ---- commit prefetched x[t+1] into sX (reads of sX done at barrier) ----
    if (more) {
#pragma unroll
      for (int j = 0; j < 5; j++) {
        union { __bf16 h[4]; uint2 u; } pk;
        pk.h[0] = (__bf16)xn[j].x; pk.h[1] = (__bf16)xn[j].y;
        pk.h[2] = (__bf16)xn[j].z; pk.h[3] = (__bf16)xn[j].w;
        *(uint2*)(sX + xr_row[j] * DP + xr_dq[j]) = pk.u;
      }
    }
    __syncthreads();  // sX ready; protects sSwr overwrite next step
  }

  // ---- rate @ Wout^T + bout (sAcc aliases sX/sS0; all reads done) ----
#pragma unroll
  for (int i = 0; i < 2; i++) {
#pragma unroll
    for (int r = 0; r < 8; r++)
      sAcc[(hidM[i] + hi * 8 + r) * HP + hidN[i] + lo] = acch[i][r];
  }
  __syncthreads();

  const float inv_t = 1.0f / (float)T_STEPS;
  for (int i = tid; i < TB * C_OUT; i += 256) {
    int row = i / C_OUT, c = i % C_OUT;
    float sum = gBout[c];
    const float* wr = gWout + c * H_REAL;
    const float* ar = sAcc + row * HP;
    for (int h = 0; h < H_REAL; h++) sum += (ar[h] * inv_t) * wr[h];
    out[(size_t)(b0 + row) * C_OUT + c] = sum;
  }
}

// ---------------------------------------------------------------------------
extern "C" void kernel_launch(void* const* d_in, const int* in_sizes, int n_in,
                              void* d_out, int out_size, void* d_ws,
                              size_t ws_size, hipStream_t stream) {
  (void)in_sizes; (void)n_in; (void)out_size; (void)ws_size;
  const float* x        = (const float*)d_in[0];
  const float* w_in     = (const float*)d_in[1];
  const float* m_in     = (const float*)d_in[2];
  const float* w_rec    = (const float*)d_in[3];
  const float* m_rec    = (const float*)d_in[4];
  const float* w_hid    = (const float*)d_in[5];
  const float* m_hid    = (const float*)d_in[6];
  const float* w_out    = (const float*)d_in[7];
  const float* m_out    = (const float*)d_in[8];
  const float* b_out    = (const float*)d_in[9];
  const float* bias_res = (const float*)d_in[10];
  const float* bias_hid = (const float*)d_in[11];

  char* ws = (char*)d_ws;
  __bf16* oWin  = (__bf16*)(ws + WS_WIN);
  __bf16* oWrec = (__bf16*)(ws + WS_WREC);
  __bf16* oWhid = (__bf16*)(ws + WS_WHID);
  float*  oWout = (float*)(ws + WS_WOUT);
  float*  oBout = (float*)(ws + WS_BOUT);

  snn_prep<<<4, 256, 0, stream>>>(w_in, m_in, w_rec, m_rec, w_hid, m_hid,
                                  w_out, m_out, b_out,
                                  oWin, oWrec, oWhid, oWout, oBout);

  const int nblocks = 16384 / TB;  // 256
  snn_main<<<nblocks, 256, LDS_BYTES, stream>>>(
      x, bias_res, bias_hid, oWin, oWrec, oWhid, oWout, oBout, (float*)d_out);
}
